// CCA_Attention_58652073394668
// MI455X (gfx1250) — matrix operand order repaired and checked by reading, not compiled, not run on hardware
//
#include <hip/hip_runtime.h>
#include <hip/hip_bf16.h>

// ---------------- problem constants ----------------
#define BB   8
#define NN   4096
#define CC   512
#define NH   8
#define HD   64
#define C3   1536
#define BN   (BB*NN)        // 32768
#define EPS  1e-12f

typedef __attribute__((ext_vector_type(16))) __bf16 v16bf;
typedef __attribute__((ext_vector_type(8)))  float  v8f;
typedef unsigned short u16;
typedef unsigned int   u32;

union Frag {            // 8 VGPRs = 16 bf16
    v16bf bf;
    u16   us[16];
    uint4 q4[2];
};
union Pack8 {           // 8 bf16 = one b128 store
    u16   us[8];
    uint4 q;
};

__device__ __forceinline__ u16 f2bf(float x) {
    u32 u = __builtin_bit_cast(u32, x);
    u32 r = (u + 0x7FFFu + ((u >> 16) & 1u)) >> 16;   // RNE
    return (u16)r;
}
__device__ __forceinline__ float bf2f(u16 h) {
    u32 u = ((u32)h) << 16;
    return __builtin_bit_cast(float, u);
}
__device__ __forceinline__ v8f wmma_bf16(const Frag& a, const Frag& b, v8f c) {
    return __builtin_amdgcn_wmma_f32_16x16x32_bf16(false, a.bf, false, b.bf,
                                                   (short)0, c, false, false);
}
__device__ __forceinline__ v8f v8f_zero() {
    v8f z; for (int i = 0; i < 8; ++i) z[i] = 0.0f; return z;
}
// A fragment: lane l -> row l%16, K = k0 + 16*(l/16) + 0..15 contiguous
__device__ __forceinline__ void load_a(Frag& f, const u16* arow, int k0, int lhi) {
    f.q4[0] = *(const uint4*)(arow + k0 + 16 * lhi);
    f.q4[1] = *(const uint4*)(arow + k0 + 16 * lhi + 8);
}
// B fragment: lane l -> col l%16, K = 16*(l/16) + 0..15 contiguous (brow pre-offset by 16*lhi)
__device__ __forceinline__ void load_b(Frag& f, const u16* brow) {
    f.q4[0] = *(const uint4*)(brow);
    f.q4[1] = *(const uint4*)(brow + 8);
}

// ---------------- prep: fp32 -> bf16 ----------------
__global__ void k_cvt(const float* __restrict__ src, u16* __restrict__ dst, int n) {
    for (int i = blockIdx.x * blockDim.x + threadIdx.x; i < n; i += gridDim.x * blockDim.x)
        dst[i] = f2bf(src[i]);
}
// w:[rows][cols] fp32 -> wt:[cols][rows] bf16
__global__ void k_transpose_w(const float* __restrict__ w, u16* __restrict__ wt,
                              int rows, int cols) {
    int i = blockIdx.x * blockDim.x + threadIdx.x;
    if (i < rows * cols) {
        int r = i / cols, c = i - r * cols;
        wt[(size_t)c * rows + r] = f2bf(w[i]);
    }
}

// ---------------- QKV GEMM: [BN,C] x [C,3C] ----------------
// block = 4 waves; wave: 32 rows x 64 cols (2x4 tiles); block tile 128x64; double-buffered
__global__ __launch_bounds__(128) void k_qkv(const u16* __restrict__ xb,
                                             const u16* __restrict__ wT,   // [3C][C]
                                             u16* __restrict__ qcm,        // [B*NH][HD][N]
                                             u16* __restrict__ kcm,
                                             u16* __restrict__ vnd) {      // [B*NH][N][HD]
    const int lane = threadIdx.x & 31, wave = threadIdx.x >> 5;
    const int l16 = lane & 15, lhi = lane >> 4;
    const int rowbase = blockIdx.x * 128 + wave * 32;  // global row (b*N+n)
    const int ccbase  = blockIdx.y * 64;               // column base (within 3C)

    const u16* arow0 = xb + (size_t)(rowbase + l16) * CC;
    const u16* arow1 = arow0 + (size_t)16 * CC;
    const u16* brows[4];
#pragma unroll
    for (int t = 0; t < 4; ++t)
        brows[t] = wT + (size_t)(ccbase + t * 16 + l16) * CC + 16 * lhi;

    v8f acc[2][4];
#pragma unroll
    for (int m = 0; m < 2; ++m)
#pragma unroll
        for (int t = 0; t < 4; ++t) acc[m][t] = v8f_zero();

    Frag a[2][2], b[2][4];
    load_a(a[0][0], arow0, 0, lhi);
    load_a(a[0][1], arow1, 0, lhi);
#pragma unroll
    for (int t = 0; t < 4; ++t) load_b(b[0][t], brows[t]);

    int cur = 0;
#pragma unroll
    for (int k0 = 0; k0 < CC - 32; k0 += 32) {
        const int nxt = cur ^ 1;
        load_a(a[nxt][0], arow0, k0 + 32, lhi);
        load_a(a[nxt][1], arow1, k0 + 32, lhi);
#pragma unroll
        for (int t = 0; t < 4; ++t) load_b(b[nxt][t], brows[t] + k0 + 32);
#pragma unroll
        for (int m = 0; m < 2; ++m)
#pragma unroll
            for (int t = 0; t < 4; ++t)
                acc[m][t] = wmma_bf16(a[cur][m], b[cur][t], acc[m][t]);
        cur = nxt;
    }
#pragma unroll
    for (int m = 0; m < 2; ++m)
#pragma unroll
        for (int t = 0; t < 4; ++t)
            acc[m][t] = wmma_bf16(a[cur][m], b[cur][t], acc[m][t]);

    // store: region 0/1 -> channel-major q/k ; region 2 -> v row-major [n][d]
    const int region = ccbase / CC;
    const int h  = (ccbase % CC) / HD;
    const int bidx = rowbase / NN;
    const int n0 = rowbase % NN;
    const int bh = bidx * NH + h;
#pragma unroll
    for (int m = 0; m < 2; ++m) {
#pragma unroll
        for (int t = 0; t < 4; ++t) {
            const int d = t * 16 + l16;                // channel within head
            if (region < 2) {
                Pack8 p;
#pragma unroll
                for (int i = 0; i < 8; ++i) p.us[i] = f2bf(acc[m][t][i]);
                u16* dst = (region == 0 ? qcm : kcm) +
                           ((size_t)bh * HD + d) * NN + n0 + m * 16 + 8 * lhi;
                *(uint4*)dst = p.q;                    // n contiguous
            } else {
#pragma unroll
                for (int i = 0; i < 8; ++i) {
                    int n = n0 + m * 16 + i + 8 * lhi;
                    vnd[((size_t)bh * NN + n) * HD + d] = f2bf(acc[m][t][i]);
                }
            }
        }
    }
}

// ---------------- inverse L2 norms of q/k rows (over N) ----------------
__global__ __launch_bounds__(256) void k_norm(const u16* __restrict__ qcm,
                                              const u16* __restrict__ kcm,
                                              float* __restrict__ rq,
                                              float* __restrict__ rk) {
    int row = blockIdx.x;                               // 0..8191 ; <4096 -> q
    const u16* src = (row < 4096 ? qcm + (size_t)row * NN
                                 : kcm + (size_t)(row - 4096) * NN);
    float s = 0.f;
    for (int i = threadIdx.x; i < NN; i += 256) { float v = bf2f(src[i]); s += v * v; }
    __shared__ float red[256];
    red[threadIdx.x] = s; __syncthreads();
    for (int off = 128; off > 0; off >>= 1) {
        if (threadIdx.x < off) red[threadIdx.x] += red[threadIdx.x + off];
        __syncthreads();
    }
    if (threadIdx.x == 0) {
        float inv = 1.0f / fmaxf(sqrtf(red[0]), EPS);
        (row < 4096 ? rq : rk)[row & 4095] = inv;
    }
}

// ---------------- attn = softmax( (q k^T) * rq * rk * temp ) ----------------
// one block per (b,h); 4 waves; wave w: rows 16w..16w+15, all 64 cols; K=N, double-buffered
__global__ __launch_bounds__(128) void k_attn(const u16* __restrict__ qcm,
                                              const u16* __restrict__ kcm,
                                              const float* __restrict__ rq,
                                              const float* __restrict__ rk,
                                              const float* __restrict__ temp,
                                              u16* __restrict__ attnb) {    // [B*NH][64][64]
    const int bh = blockIdx.x;
    const int lane = threadIdx.x & 31, wave = threadIdx.x >> 5;
    const int l16 = lane & 15, lhi = lane >> 4;
    const u16* qb = qcm + (size_t)bh * HD * NN;
    const u16* kb = kcm + (size_t)bh * HD * NN;

    const u16* arow = qb + (size_t)(wave * 16 + l16) * NN;
    const u16* brows[4];
#pragma unroll
    for (int t = 0; t < 4; ++t)
        brows[t] = kb + (size_t)(t * 16 + l16) * NN + 16 * lhi;

    v8f acc[4] = { v8f_zero(), v8f_zero(), v8f_zero(), v8f_zero() };
    Frag a[2], b[2][4];
    load_a(a[0], arow, 0, lhi);
#pragma unroll
    for (int t = 0; t < 4; ++t) load_b(b[0][t], brows[t]);

    int cur = 0;
    for (int n0 = 0; n0 < NN - 32; n0 += 32) {
        const int nxt = cur ^ 1;
        load_a(a[nxt], arow, n0 + 32, lhi);
#pragma unroll
        for (int t = 0; t < 4; ++t) load_b(b[nxt][t], brows[t] + n0 + 32);
#pragma unroll
        for (int t = 0; t < 4; ++t) acc[t] = wmma_bf16(a[cur], b[cur][t], acc[t]);
        cur = nxt;
    }
#pragma unroll
    for (int t = 0; t < 4; ++t) acc[t] = wmma_bf16(a[cur], b[cur][t], acc[t]);

    __shared__ float S[64][65];
    const float tscale = temp[bh & (NH - 1)];
#pragma unroll
    for (int t = 0; t < 4; ++t) {
        int d = t * 16 + l16;
        float rkd = rk[bh * HD + d] * tscale;
#pragma unroll
        for (int i = 0; i < 8; ++i) {
            int c = wave * 16 + i + 8 * lhi;
            S[c][d] = acc[t][i] * rq[bh * HD + c] * rkd;
        }
    }
    __syncthreads();
    if (threadIdx.x < 64) {
        int c = threadIdx.x;
        float mx = -3.4e38f;
        for (int d = 0; d < 64; ++d) mx = fmaxf(mx, S[c][d]);
        float sum = 0.f;
        for (int d = 0; d < 64; ++d) sum += __expf(S[c][d] - mx);
        float inv = 1.0f / sum;
        u16* dst = attnb + ((size_t)bh * 64 + c) * 64;
        for (int d = 0; d < 64; ++d) dst[d] = f2bf(__expf(S[c][d] - mx) * inv);
    }
}

// ---------------- ctx = attn @ v -> [b][n][c] bf16 ----------------
// grid (N/64, B*NH); wave handles 16 cols (n), 4 row-tiles (c), K=64 (all loads hoisted)
__global__ __launch_bounds__(128) void k_ctx(const u16* __restrict__ attnb,
                                             const u16* __restrict__ vnd,
                                             u16* __restrict__ ctx) {       // [B][N][C]
    const int bh = blockIdx.y;
    const int lane = threadIdx.x & 31, wave = threadIdx.x >> 5;
    const int l16 = lane & 15, lhi = lane >> 4;
    const u16* ab = attnb + (size_t)bh * 64 * 64;
    const u16* vb = vnd + (size_t)bh * NN * HD;
    const int ncol = blockIdx.x * 64 + wave * 16 + l16;    // this lane's B column (n)

    Frag bfr[2], afr[2][4];
#pragma unroll
    for (int s = 0; s < 2; ++s) {
        const int k0 = s * 32;
        load_b(bfr[s], vb + (size_t)ncol * HD + k0 + 16 * lhi);
#pragma unroll
        for (int t = 0; t < 4; ++t)
            load_a(afr[s][t], ab + (size_t)(t * 16 + l16) * 64, k0, lhi);
    }
    v8f acc[4] = { v8f_zero(), v8f_zero(), v8f_zero(), v8f_zero() };
#pragma unroll
    for (int s = 0; s < 2; ++s)
#pragma unroll
        for (int t = 0; t < 4; ++t)
            acc[t] = wmma_bf16(afr[s][t], bfr[s], acc[t]);

    const int bidx = bh >> 3, h = bh & 7;
#pragma unroll
    for (int t = 0; t < 4; ++t) {
        Pack8 p;
#pragma unroll
        for (int i = 0; i < 8; ++i) p.us[i] = f2bf(acc[t][i]);
        u16* dst = ctx + ((size_t)(bidx * NN + ncol)) * CC + h * HD + t * 16 + 8 * lhi;
        *(uint4*)dst = p.q;                                // c contiguous
    }
}

// ---------------- out = ctx @ Wproj + bias (fp32) ----------------
// same pipelined 32x64-per-wave structure as k_qkv
__global__ __launch_bounds__(128) void k_proj(const u16* __restrict__ ctx,
                                              const u16* __restrict__ wT,   // [C][C]
                                              const float* __restrict__ bias,
                                              float* __restrict__ out) {
    const int lane = threadIdx.x & 31, wave = threadIdx.x >> 5;
    const int l16 = lane & 15, lhi = lane >> 4;
    const int rowbase = blockIdx.x * 128 + wave * 32;
    const int colbase = blockIdx.y * 64;

    const u16* arow0 = ctx + (size_t)(rowbase + l16) * CC;
    const u16* arow1 = arow0 + (size_t)16 * CC;
    const u16* brows[4];
#pragma unroll
    for (int t = 0; t < 4; ++t)
        brows[t] = wT + (size_t)(colbase + t * 16 + l16) * CC + 16 * lhi;

    v8f acc[2][4];
#pragma unroll
    for (int m = 0; m < 2; ++m)
#pragma unroll
        for (int t = 0; t < 4; ++t) acc[m][t] = v8f_zero();

    Frag a[2][2], b[2][4];
    load_a(a[0][0], arow0, 0, lhi);
    load_a(a[0][1], arow1, 0, lhi);
#pragma unroll
    for (int t = 0; t < 4; ++t) load_b(b[0][t], brows[t]);

    int cur = 0;
#pragma unroll
    for (int k0 = 0; k0 < CC - 32; k0 += 32) {
        const int nxt = cur ^ 1;
        load_a(a[nxt][0], arow0, k0 + 32, lhi);
        load_a(a[nxt][1], arow1, k0 + 32, lhi);
#pragma unroll
        for (int t = 0; t < 4; ++t) load_b(b[nxt][t], brows[t] + k0 + 32);
#pragma unroll
        for (int m = 0; m < 2; ++m)
#pragma unroll
            for (int t = 0; t < 4; ++t)
                acc[m][t] = wmma_bf16(a[cur][m], b[cur][t], acc[m][t]);
        cur = nxt;
    }
#pragma unroll
    for (int m = 0; m < 2; ++m)
#pragma unroll
        for (int t = 0; t < 4; ++t)
            acc[m][t] = wmma_bf16(a[cur][m], b[cur][t], acc[m][t]);

#pragma unroll
    for (int m = 0; m < 2; ++m) {
#pragma unroll
        for (int t = 0; t < 4; ++t) {
            int col = colbase + t * 16 + l16;
            float bv = bias[col];
#pragma unroll
            for (int i = 0; i < 8; ++i) {
                int r = rowbase + m * 16 + i + 8 * lhi;
                out[(size_t)r * CC + col] = acc[m][t][i] + bv;
            }
        }
    }
}

// ---------------- host launch ----------------
extern "C" void kernel_launch(void* const* d_in, const int* in_sizes, int n_in,
                              void* d_out, int out_size, void* d_ws, size_t ws_size,
                              hipStream_t stream) {
    const float* x      = (const float*)d_in[0];
    // d_in[1] = x_out (unused by reference)
    const float* W_qkv  = (const float*)d_in[2];
    const float* temp   = (const float*)d_in[3];
    const float* W_proj = (const float*)d_in[4];
    const float* b_proj = (const float*)d_in[5];
    float* out = (float*)d_out;

    char* ws = (char*)d_ws;
    const size_t SZ_XB  = (size_t)BN * CC * 2;            // 32 MB (reused as ctx)
    const size_t SZ_HCM = (size_t)BB * NH * HD * NN * 2;  // 32 MB
    u16* xb     = (u16*)(ws);                             // [BN][C]   (later: ctx)
    u16* qcm    = (u16*)(ws + SZ_XB);                     // [B*NH][HD][N]
    u16* kcm    = (u16*)(ws + SZ_XB + SZ_HCM);
    u16* vnd    = (u16*)(ws + SZ_XB + 2 * SZ_HCM);        // [B*NH][N][HD]
    char* p     = ws + SZ_XB + 3 * SZ_HCM;
    u16* wqkvT  = (u16*)p;            p += (size_t)C3 * CC * 2;   // [3C][C]
    u16* wprojT = (u16*)p;            p += (size_t)CC * CC * 2;   // [C][C]
    u16* attnb  = (u16*)p;            p += (size_t)BB * NH * 64 * 64 * 2;
    float* rq   = (float*)p;          p += 4096 * 4;
    float* rk   = (float*)p;          p += 4096 * 4;
    u16* ctx    = xb;                                     // reuse xb buffer

    // 1) convert / transpose to bf16
    k_cvt<<<8192, 256, 0, stream>>>(x, xb, BN * CC);
    k_transpose_w<<<(CC * C3 + 255) / 256, 256, 0, stream>>>(W_qkv, wqkvT, CC, C3);
    k_transpose_w<<<(CC * CC + 255) / 256, 256, 0, stream>>>(W_proj, wprojT, CC, CC);
    // 2) QKV GEMM -> q/k channel-major, v row-major
    k_qkv<<<dim3(BN / 128, C3 / 64), 128, 0, stream>>>(xb, wqkvT, qcm, kcm, vnd);
    // 3) inverse L2 norms over N
    k_norm<<<8192, 256, 0, stream>>>(qcm, kcm, rq, rk);
    // 4) Gram + scale + softmax
    k_attn<<<BB * NH, 128, 0, stream>>>(qcm, kcm, rq, rk, temp, attnb);
    // 5) ctx = attn @ v  (written [b][n][c])
    k_ctx<<<dim3(NN / 64, BB * NH), 128, 0, stream>>>(attnb, vnd, ctx);
    // 6) projection + bias -> fp32 out
    k_proj<<<dim3(BN / 128, CC / 64), 128, 0, stream>>>(ctx, wprojT, b_proj, out);
}
